// PETRMultiheadAttention_12455405158453
// MI455X (gfx1250) — matrix operand order repaired and checked by reading, not compiled, not run on hardware
//
#include <hip/hip_runtime.h>

// Problem constants (match reference)
#define E_   256
#define NH   8
#define HD   32
#define TGT_ 1024
#define SRC_ 4096
#define BB   4

typedef __attribute__((ext_vector_type(16))) _Float16 v16h;
typedef __attribute__((ext_vector_type(8)))  _Float16 h8v;
typedef __attribute__((ext_vector_type(8)))  float    v8f;
typedef __attribute__((ext_vector_type(4)))  float    f4;

static __device__ __forceinline__ v8f wmma16(v16h a, v16h b, v8f c) {
  // D = A(16x32 f16) x B(32x16 f16) + C(16x16 f32)
  return __builtin_amdgcn_wmma_f32_16x16x32_f16(false, a, false, b, (short)0, c,
                                                false, false);
}

// Build A fragment (elems 0..7 from p0[0..7], 8..15 from p1[0..7]) from f32.
static __device__ __forceinline__ v16h fragA_f32(const float* p0, const float* p1) {
  union { v16h v; _Float16 e[16]; } u;
  f4 a0 = *(const f4*)p0, a1 = *(const f4*)(p0 + 4);
  f4 b0 = *(const f4*)p1, b1 = *(const f4*)(p1 + 4);
  #pragma unroll
  for (int i = 0; i < 4; i++) {
    u.e[i]      = (_Float16)a0[i];
    u.e[4 + i]  = (_Float16)a1[i];
    u.e[8 + i]  = (_Float16)b0[i];
    u.e[12 + i] = (_Float16)b1[i];
  }
  return u.v;
}

// Build B fragment from 16 contiguous f32 at p.
static __device__ __forceinline__ v16h fragB_f32(const float* p) {
  union { v16h v; _Float16 e[16]; } u;
  f4 c0 = *(const f4*)p,       c1 = *(const f4*)(p + 4);
  f4 c2 = *(const f4*)(p + 8), c3 = *(const f4*)(p + 12);
  #pragma unroll
  for (int i = 0; i < 4; i++) {
    u.e[i]      = (_Float16)c0[i];
    u.e[4 + i]  = (_Float16)c1[i];
    u.e[8 + i]  = (_Float16)c2[i];
    u.e[12 + i] = (_Float16)c3[i];
  }
  return u.v;
}

// ---------------------------------------------------------------------------
// zero attn_w region of d_out (accumulated with f32 atomics later)
// ---------------------------------------------------------------------------
__global__ void zero_kernel(f4* __restrict__ p, size_t n4) {
  size_t i = (size_t)blockIdx.x * blockDim.x + threadIdx.x;
  if (i < n4) p[i] = (f4){0.f, 0.f, 0.f, 0.f};
}

// ---------------------------------------------------------------------------
// Fused input projection: C = ((X [+ Pos]) @ W_sub^T + b_sub) * oscale, f16.
// GEMM rows batch-major: m = b*rows + r, memory row = X[(r*BB + b)*E].
// MODE 0: Q -> out[(b*NH+h)*rows + r][d]  (oscale = 1/sqrt(HD), pre-scales QK^T)
// MODE 1: K (same layout, rows=SRC)       MODE 2: V -> out[(b*NH+h)*HD+d][s]
// One wave per 16x16 output tile; K-loop = 8 steps of 32.
// ---------------------------------------------------------------------------
template <bool HAS_POS, int MODE>
__global__ __launch_bounds__(256) void proj_kernel(
    const float* __restrict__ X, const float* __restrict__ Pos,
    const float* __restrict__ W, const float* __restrict__ bias,
    _Float16* __restrict__ out, int rows, float oscale)
{
  __shared__ _Float16 tbuf[8][16][17];  // per-wave transpose tile (MODE 2)

  const int lane  = threadIdx.x & 31;
  const int wid   = threadIdx.x >> 5;
  const int gw    = blockIdx.x * 8 + wid;
  const int mtile = gw >> 4;            // 16 n-tiles (N = 256)
  const int ntile = gw & 15;
  const int m0 = mtile * 16, n0 = ntile * 16;
  const int am = lane & 15;             // A row / B col / D col within tile
  const int hb = lane >> 4;             // K-half selector
  const int b  = m0 / rows;             // batch (tiles never straddle batches)
  const int r0 = m0 - b * rows;         // sequence-row base of tile

  const float* xrow = X + ((size_t)(r0 + am) * BB + b) * E_;
  const float* prow = Pos + ((size_t)(r0 + am) * BB + b) * E_;
  const float* wcol = W + (size_t)(n0 + am) * E_;   // B col n = W row f

  v8f c = {0.f,0.f,0.f,0.f,0.f,0.f,0.f,0.f};
  #pragma unroll
  for (int k0 = 0; k0 < E_; k0 += 32) {
    v16h a;
    if (HAS_POS) {
      // x + pos, vector loads, then pack-convert to f16
      union { v16h v; _Float16 e[16]; } u;
      const float* xp0 = xrow + k0 + hb * 8;
      const float* xp1 = xrow + k0 + 16 + hb * 8;
      const float* pp0 = prow + k0 + hb * 8;
      const float* pp1 = prow + k0 + 16 + hb * 8;
      f4 a0 = *(const f4*)xp0 + *(const f4*)pp0;
      f4 a1 = *(const f4*)(xp0 + 4) + *(const f4*)(pp0 + 4);
      f4 b0 = *(const f4*)xp1 + *(const f4*)pp1;
      f4 b1 = *(const f4*)(xp1 + 4) + *(const f4*)(pp1 + 4);
      #pragma unroll
      for (int i = 0; i < 4; i++) {
        u.e[i]      = (_Float16)a0[i];
        u.e[4 + i]  = (_Float16)a1[i];
        u.e[8 + i]  = (_Float16)b0[i];
        u.e[12 + i] = (_Float16)b1[i];
      }
      a = u.v;
    } else {
      a = fragA_f32(xrow + k0 + hb * 8, xrow + k0 + 16 + hb * 8);
    }
    v16h bf = fragA_f32(wcol + k0 + hb * 8, wcol + k0 + 16 + hb * 8);
    c = wmma16(a, bf, c);
  }

  const int   f  = n0 + am;
  const float bv = bias[f];

  if (MODE != 2) {
    const int h = f >> 5, d = f & 31;
    #pragma unroll
    for (int j = 0; j < 8; j++) {               // D elem j -> row j + 8*hb
      const int r = r0 + j + 8 * hb;
      out[((size_t)(b * NH + h) * rows + r) * HD + d] =
          (_Float16)((c[j] + bv) * oscale);
    }
  } else {
    // V: transpose tile through LDS so [d][s] stores are 32B contiguous
    #pragma unroll
    for (int j = 0; j < 8; j++)
      tbuf[wid][j + 8 * hb][am] = (_Float16)(c[j] + bv);
    __syncthreads();
    if (lane < 16) {
      v16h pk;
      #pragma unroll
      for (int mm = 0; mm < 16; mm++) pk[mm] = tbuf[wid][mm][lane];
      const int h = n0 >> 5, dbase = n0 & 31;
      const size_t idx =
          ((size_t)(b * NH + h) * HD + dbase + lane) * (size_t)SRC_ + r0;
      *(v16h*)(out + idx) = pk;                 // 32-byte aligned store
    }
  }
}

// ---------------------------------------------------------------------------
// Attention: one WG per (b, h, 16-row tgt tile). 512 threads = 16 waves.
// LDS: S[16][4096] f32 score strip (256KB fits 320KB WGP LDS) + row inv-sums
// + 16 per-wave partial ctx tiles. Scores never touch HBM.
// Q was pre-scaled by 1/sqrt(HD) in the projection, so score WMMA results go
// straight to LDS with no post-scaling VALU work.
// ---------------------------------------------------------------------------
__global__ __launch_bounds__(512) void attn_kernel(
    const _Float16* __restrict__ q16, const _Float16* __restrict__ k16,
    const _Float16* __restrict__ vt, float* __restrict__ ctx,
    float* __restrict__ attnw)
{
  extern __shared__ float smem[];
  float* S     = smem;                    // 16*SRC_
  float* rinv  = S + 16 * SRC_;           // 16
  float* credw = rinv + 16;               // 16 waves * 512

  const int tid = threadIdx.x, lane = tid & 31, wid = tid >> 5;
  const int am = lane & 15, hb = lane >> 4;
  const int t0 = blockIdx.x * 16;
  const int bh = blockIdx.z * NH + blockIdx.y;

  // Q fragment (A operand): rows t0..t0+15, K = head dim (exactly 32)
  const _Float16* qrow = q16 + ((size_t)bh * TGT_ + t0 + am) * HD;
  union { v16h v; h8v h[2]; } uq;
  uq.h[0] = *(const h8v*)(qrow + hb * 16);
  uq.h[1] = *(const h8v*)(qrow + hb * 16 + 8);
  const v16h aq = uq.v;

  // ---- scores: each wave owns a 256-column chunk of SRC ----
  const int sbase = wid * (SRC_ / 16);
  for (int st = 0; st < SRC_ / 16; st += 16) {
    const int s0 = sbase + st;
    // B col n = key row s0+am (contiguous along d); 32B vector load
    const v16h bk =
        *(const v16h*)(k16 + ((size_t)bh * SRC_ + s0 + am) * HD + hb * 16);
    v8f c = {0.f,0.f,0.f,0.f,0.f,0.f,0.f,0.f};
    c = wmma16(aq, bk, c);                 // one WMMA per 16 key columns
    #pragma unroll
    for (int j = 0; j < 8; j++)
      S[(j + 8 * hb) * SRC_ + s0 + am] = c[j];
  }
  __syncthreads();

  // ---- softmax: wave w owns row w (16 waves == 16 rows) ----
  {
    float* Sr = S + wid * SRC_;
    float mx = -3.4e38f;
    for (int cix = lane * 4; cix < SRC_; cix += 128) {
      f4 v = *(const f4*)(Sr + cix);
      mx = fmaxf(mx, fmaxf(fmaxf(v.x, v.y), fmaxf(v.z, v.w)));
    }
    #pragma unroll
    for (int off = 16; off > 0; off >>= 1)
      mx = fmaxf(mx, __shfl_xor(mx, off, 32));
    float sum = 0.f;
    for (int cix = lane * 4; cix < SRC_; cix += 128) {
      f4 v = *(const f4*)(Sr + cix);
      f4 e = {__expf(v.x - mx), __expf(v.y - mx),
              __expf(v.z - mx), __expf(v.w - mx)};
      *(f4*)(Sr + cix) = e;                // keep unnormalized exp in LDS
      sum += e.x + e.y + e.z + e.w;
    }
    #pragma unroll
    for (int off = 16; off > 0; off >>= 1)
      sum += __shfl_xor(sum, off, 32);
    if (lane == 0) rinv[wid] = 1.f / sum;
  }
  __syncthreads();

  // ---- attn_w head-mean: f32 atomics, target is 67MB < 192MB L2 ----
  {
    const float wnorm = rinv[wid] * 0.125f;  // 1/H
    float* aw = attnw + ((size_t)blockIdx.z * TGT_ + t0 + wid) * SRC_;
    const float* Sr = S + wid * SRC_;
    for (int cix = lane * 4; cix < SRC_; cix += 128) {
      f4 v = *(const f4*)(Sr + cix);
      unsafeAtomicAdd(&aw[cix],     v.x * wnorm);
      unsafeAtomicAdd(&aw[cix + 1], v.y * wnorm);
      unsafeAtomicAdd(&aw[cix + 2], v.z * wnorm);
      unsafeAtomicAdd(&aw[cix + 3], v.w * wnorm);
    }
  }

  // ---- P @ V: each wave accumulates its 256-col K-chunk (8 WMMA pairs) ----
  v8f c0 = {0.f,0.f,0.f,0.f,0.f,0.f,0.f,0.f};
  v8f c1 = {0.f,0.f,0.f,0.f,0.f,0.f,0.f,0.f};
  const float rs = rinv[am];               // A-fragment row scale (row m = am)
  for (int kk = 0; kk < SRC_ / 16; kk += 32) {
    const int ks = sbase + kk;
    // A fragment: probabilities from LDS (b128 reads), scaled, f16-converted
    union { v16h v; _Float16 e[16]; } up;
    const float* sp0 = S + am * SRC_ + ks + hb * 16;
    const float* sp1 = S + am * SRC_ + ks + hb * 16 + 8;
    f4 a0 = *(const f4*)sp0, a1 = *(const f4*)(sp0 + 4);
    f4 b0 = *(const f4*)sp1, b1 = *(const f4*)(sp1 + 4);
    #pragma unroll
    for (int i = 0; i < 4; i++) {
      up.e[i]      = (_Float16)(a0[i] * rs);
      up.e[4 + i]  = (_Float16)(a1[i] * rs);
      up.e[8 + i]  = (_Float16)(b0[i] * rs);
      up.e[12 + i] = (_Float16)(b1[i] * rs);
    }
    // B fragments: vt rows are contiguous along s -> 32B vector loads
    const v16h bv0 =
        *(const v16h*)(vt + ((size_t)bh * HD + am) * SRC_ + ks + hb * 16);
    const v16h bv1 =
        *(const v16h*)(vt + ((size_t)bh * HD + 16 + am) * SRC_ + ks + hb * 16);
    c0 = wmma16(up.v, bv0, c0);
    c1 = wmma16(up.v, bv1, c1);
  }
  #pragma unroll
  for (int j = 0; j < 8; j++) {            // per-wave partial 16x32 tile
    credw[wid * 512 + (j + 8 * hb) * HD + am]      = c0[j];
    credw[wid * 512 + (j + 8 * hb) * HD + 16 + am] = c1[j];
  }
  __syncthreads();
  {
    const int m = tid >> 5, d = tid & 31;  // 512 threads, 512 outputs
    float acc = 0.f;
    #pragma unroll
    for (int w = 0; w < 16; w++) acc += credw[w * 512 + m * HD + d];
    ctx[((size_t)bh * TGT_ + t0 + m) * HD + d] = acc;
  }
}

// ---------------------------------------------------------------------------
// Output projection + bias + residual: out = ctx @ Wo^T + bo + query
// ctx layout [b][h][t][d]; K-steps of 32 align exactly with head blocks.
// ---------------------------------------------------------------------------
__global__ __launch_bounds__(256) void outproj_kernel(
    const float* __restrict__ ctx, const float* __restrict__ Wo,
    const float* __restrict__ bo, const float* __restrict__ identity,
    float* __restrict__ out)
{
  const int lane = threadIdx.x & 31, wid = threadIdx.x >> 5;
  const int gw = blockIdx.x * 8 + wid;
  const int mtile = gw >> 4, ntile = gw & 15;
  const int m0 = mtile * 16, n0 = ntile * 16;
  const int am = lane & 15, hb = lane >> 4;
  const int b  = m0 >> 10;                 // TGT = 1024 rows per batch
  const int t0 = m0 & (TGT_ - 1);

  const float* wrow = Wo + (size_t)(n0 + am) * E_;
  v8f c = {0.f,0.f,0.f,0.f,0.f,0.f,0.f,0.f};
  #pragma unroll
  for (int h = 0; h < NH; h++) {           // k0 = h*32 (one head per K-step)
    const float* arow = ctx + ((size_t)(b * NH + h) * TGT_ + t0 + am) * HD;
    v16h a  = fragA_f32(arow + hb * 8, arow + 16 + hb * 8);
    v16h bf = fragA_f32(wrow + h * 32 + hb * 8, wrow + h * 32 + 16 + hb * 8);
    c = wmma16(a, bf, c);
  }
  const int   f  = n0 + am;
  const float bv = bo[f];
  #pragma unroll
  for (int j = 0; j < 8; j++) {
    const int t = t0 + j + 8 * hb;
    const size_t oi = ((size_t)t * BB + b) * E_ + f;
    out[oi] = c[j] + bv + identity[oi];
  }
}

// ---------------------------------------------------------------------------
extern "C" void kernel_launch(void* const* d_in, const int* in_sizes, int n_in,
                              void* d_out, int out_size, void* d_ws, size_t ws_size,
                              hipStream_t stream) {
  const float* query = (const float*)d_in[0];
  const float* key   = (const float*)d_in[1];
  const float* value = (const float*)d_in[2];
  const float* qpos  = (const float*)d_in[3];
  const float* kpos  = (const float*)d_in[4];
  const float* ipw   = (const float*)d_in[5];
  const float* ipb   = (const float*)d_in[6];
  const float* opw   = (const float*)d_in[7];
  const float* opb   = (const float*)d_in[8];

  float* out   = (float*)d_out;                         // (TGT,B,E)
  float* attnw = out + (size_t)TGT_ * BB * E_;          // (B,TGT,SRC)

  // Workspace: q16 (2MB) | k16 (8MB) | vt (8MB) | ctx f32 (4MB)  = 22MB
  _Float16* q16 = (_Float16*)d_ws;
  _Float16* k16 = q16 + (size_t)BB * NH * TGT_ * HD;
  _Float16* vt  = k16 + (size_t)BB * NH * SRC_ * HD;
  float*    ctx = (float*)(vt + (size_t)BB * NH * SRC_ * HD);

  const size_t nz4 = (size_t)BB * TGT_ * SRC_ / 4;
  zero_kernel<<<(unsigned)((nz4 + 255) / 256), 256, 0, stream>>>((f4*)attnw, nz4);

  const float qscale = 0.1767766952966369f;  // 1/sqrt(HD), folded into Q proj
  // 8 waves/block, 1 wave per 16x16 output tile
  proj_kernel<true, 0><<<512,  256, 0, stream>>>(query, qpos, ipw,            ipb,          q16, TGT_, qscale);
  proj_kernel<true, 1><<<2048, 256, 0, stream>>>(key,   kpos, ipw + E_ * E_,  ipb + E_,     k16, SRC_, 1.0f);
  proj_kernel<false,2><<<2048, 256, 0, stream>>>(value, nullptr, ipw + 2 * E_ * E_, ipb + 2 * E_, vt, SRC_, 1.0f);

  const size_t smem_bytes =
      (size_t)(16 * SRC_ + 16 + 16 * 512) * sizeof(float);  // ~288KB < 320KB
  attn_kernel<<<dim3(TGT_ / 16, NH, BB), 512, smem_bytes, stream>>>(
      q16, k16, vt, ctx, attnw);

  outproj_kernel<<<512, 256, 0, stream>>>(ctx, opw, opb, query, out);
}
